// BitNetLinear_89489938580100
// MI455X (gfx1250) — compile-verified, matrix operand order checked by reading, not run on hardware
//
#include <hip/hip_runtime.h>
#include <stdint.h>

typedef __bf16 v16bf __attribute__((ext_vector_type(16)));
typedef __bf16 v8bf  __attribute__((ext_vector_type(8)));
typedef float  v8f   __attribute__((ext_vector_type(8)));
typedef float  v4f   __attribute__((ext_vector_type(4)));
typedef int    v4i   __attribute__((ext_vector_type(4)));
typedef unsigned short ushort8 __attribute__((ext_vector_type(8)));

#if __has_builtin(__builtin_amdgcn_global_load_async_to_lds_b128) && \
    __has_builtin(__builtin_amdgcn_s_wait_asynccnt)
#define USE_ASYNC 1
#else
#define USE_ASYNC 0
#endif

#define BM 256         // block rows (4 waves x 64)
#define BN 128         // block cols (2 waves x 64)
#define BK 32          // K-slice per LDS tile (bf16 elements)
#define BKP 40         // padded row stride in halfs: 80B, 16B-aligned, conflict-free
#define THREADS 256    // 8 wave32 waves

// ---------- scalar converters ----------
__device__ __forceinline__ unsigned short f32_to_bf16(float f) {
  unsigned int u = __builtin_bit_cast(unsigned int, f);
  unsigned int r = (u + 0x7FFFu + ((u >> 16) & 1u)) >> 16;   // RNE
  return (unsigned short)r;
}
__device__ __forceinline__ unsigned short sign_bf16(float w) {
  return w > 0.0f ? (unsigned short)0x3F80u
       : (w < 0.0f ? (unsigned short)0xBF80u : (unsigned short)0u);
}

// ---------- pass 1: fp32 -> bf16 conversions into workspace ----------
__global__ void cvt_x_kernel(const float* __restrict__ x,
                             unsigned short* __restrict__ xb, long long n) {
  long long i = ((long long)blockIdx.x * blockDim.x + threadIdx.x) * 8;
  if (i + 8 > n) return;
  v4f a = __builtin_nontemporal_load((const v4f*)(x + i));
  v4f b = __builtin_nontemporal_load((const v4f*)(x + i + 4));
  ushort8 o;
#pragma unroll
  for (int t = 0; t < 4; ++t) { o[t] = f32_to_bf16(a[t]); o[t + 4] = f32_to_bf16(b[t]); }
  *(ushort8*)(xb + i) = o;
}

__global__ void cvt_w_kernel(const float* __restrict__ w,
                             unsigned short* __restrict__ wb, long long n) {
  long long i = ((long long)blockIdx.x * blockDim.x + threadIdx.x) * 8;
  if (i + 8 > n) return;
  v4f a = __builtin_nontemporal_load((const v4f*)(w + i));
  v4f b = __builtin_nontemporal_load((const v4f*)(w + i + 4));
  ushort8 o;
#pragma unroll
  for (int t = 0; t < 4; ++t) { o[t] = sign_bf16(a[t]); o[t + 4] = sign_bf16(b[t]); }
  *(ushort8*)(wb + i) = o;
}

// ---------- pass 2: bf16 WMMA GEMM  out = Xq @ Wq^T + bias ----------
// Block 256x128, 8 waves, each wave owns a 64x64 tile (4x4 WMMA 16x16 tiles).
// PRECONV=1: read bf16 from workspace (async-copy to LDS when available).
// PRECONV=0: read fp32 from inputs, convert on the fly while filling LDS.
template <int PRECONV>
__global__ __launch_bounds__(THREADS)
void bitnet_gemm(const float* __restrict__ X, const float* __restrict__ W,
                 const unsigned short* __restrict__ Xb,
                 const unsigned short* __restrict__ Wb,
                 const float* __restrict__ bias, float* __restrict__ out,
                 int M, int N, int K) {
  __shared__ unsigned short As[2][BM][BKP];   // 40 KB
  __shared__ unsigned short Bs[2][BN][BKP];   // 20 KB

  const int tid   = threadIdx.x;
  const int lane  = tid & 31;
  const int wm    = (tid >> 5) >> 1;   // 0..3  (wave row, 64 rows each)
  const int wn    = (tid >> 5) & 1;    // 0..1  (wave col, 64 cols each)
  const int m_blk = blockIdx.y * BM;
  const int n_blk = blockIdx.x * BN;
  const int hi    = lane >> 4;         // 0/1 half-wave
  const int l15   = lane & 15;
  const int kb    = hi << 3;           // K base: 0 or 8 (per ISA A/B layout)

  // accumulators: 4x4 tiles of 16x16, bias-initialized (C layout: N = lane&15)
  v8f acc[4][4];
#pragma unroll
  for (int j = 0; j < 4; ++j) {
    const float bv = bias[n_blk + wn * 64 + j * 16 + l15];
    v8f t = {bv, bv, bv, bv, bv, bv, bv, bv};
#pragma unroll
    for (int i = 0; i < 4; ++i) acc[i][j] = t;
  }

  // fill-stage chunk coords: A has 1024 16B-chunks (4/thread), B has 512 (2/thread)
  int rA[4], cA[4], rB[2], cB[2];
  unsigned offA[4], offB[2];           // 32-bit element offsets -> saddr+voffset
#pragma unroll
  for (int i = 0; i < 4; ++i) {
    const int id = tid + i * THREADS;  // 0..1023
    rA[i] = id >> 2;                   // row 0..255
    cA[i] = (id & 3) * 8;              // half-col 0,8,16,24
    offA[i] = (unsigned)(m_blk + rA[i]) * (unsigned)K + (unsigned)cA[i];
  }
#pragma unroll
  for (int i = 0; i < 2; ++i) {
    const int id = tid + i * THREADS;  // 0..511
    rB[i] = id >> 2;                   // row 0..127
    cB[i] = (id & 3) * 8;
    offB[i] = (unsigned)(n_blk + rB[i]) * (unsigned)K + (unsigned)cB[i];
  }

#if USE_ASYNC
  auto fill_async = [&](int buf, int kt) {
    const unsigned kk = (unsigned)kt * BK;
#pragma unroll
    for (int i = 0; i < 4; ++i)
      __builtin_amdgcn_global_load_async_to_lds_b128(
          (v4i*)(Xb + (size_t)(offA[i] + kk)),
          (v4i*)&As[buf][rA[i]][cA[i]], 0, 0);
#pragma unroll
    for (int i = 0; i < 2; ++i)
      __builtin_amdgcn_global_load_async_to_lds_b128(
          (v4i*)(Wb + (size_t)(offB[i] + kk)),
          (v4i*)&Bs[buf][rB[i]][cB[i]], 0, 0);
  };
#endif
  auto fill_sync_pre = [&](int buf, int kt) {
    const unsigned kk = (unsigned)kt * BK;
#pragma unroll
    for (int i = 0; i < 4; ++i)
      *(ushort8*)&As[buf][rA[i]][cA[i]] = *(const ushort8*)(Xb + (size_t)(offA[i] + kk));
#pragma unroll
    for (int i = 0; i < 2; ++i)
      *(ushort8*)&Bs[buf][rB[i]][cB[i]] = *(const ushort8*)(Wb + (size_t)(offB[i] + kk));
  };
  auto fill_fused = [&](int buf, int kt) {
    const unsigned kk = (unsigned)kt * BK;
#pragma unroll
    for (int i = 0; i < 4; ++i) {
      const float* pa = X + (size_t)(offA[i] + kk);
      v4f a0 = *(const v4f*)pa, a1 = *(const v4f*)(pa + 4);
      ushort8 ua;
#pragma unroll
      for (int t = 0; t < 4; ++t) { ua[t] = f32_to_bf16(a0[t]); ua[t + 4] = f32_to_bf16(a1[t]); }
      *(ushort8*)&As[buf][rA[i]][cA[i]] = ua;
    }
#pragma unroll
    for (int i = 0; i < 2; ++i) {
      const float* pb = W + (size_t)(offB[i] + kk);
      v4f b0 = *(const v4f*)pb, b1 = *(const v4f*)(pb + 4);
      ushort8 ub;
#pragma unroll
      for (int t = 0; t < 4; ++t) { ub[t] = sign_bf16(b0[t]); ub[t + 4] = sign_bf16(b1[t]); }
      *(ushort8*)&Bs[buf][rB[i]][cB[i]] = ub;
    }
  };

  auto compute = [&](int buf) {
    v16bf a[4], b[4];
#pragma unroll
    for (int i = 0; i < 4; ++i) {
      const int r = wm * 64 + i * 16 + l15;
      v8bf lo = *(const v8bf*)&As[buf][r][kb];
      v8bf hi8 = *(const v8bf*)&As[buf][r][kb + 16];
      a[i] = __builtin_shufflevector(lo, hi8, 0, 1, 2, 3, 4, 5, 6, 7,
                                     8, 9, 10, 11, 12, 13, 14, 15);
    }
#pragma unroll
    for (int j = 0; j < 4; ++j) {
      const int r = wn * 64 + j * 16 + l15;   // B column = W row (K-contiguous)
      v8bf lo = *(const v8bf*)&Bs[buf][r][kb];
      v8bf hi8 = *(const v8bf*)&Bs[buf][r][kb + 16];
      b[j] = __builtin_shufflevector(lo, hi8, 0, 1, 2, 3, 4, 5, 6, 7,
                                     8, 9, 10, 11, 12, 13, 14, 15);
    }
#pragma unroll
    for (int i = 0; i < 4; ++i)
#pragma unroll
      for (int j = 0; j < 4; ++j)
        acc[i][j] = __builtin_amdgcn_wmma_f32_16x16x32_bf16(
            false, a[i], false, b[j], (short)0, acc[i][j], false, false);
  };

  const int NT = K / BK;
  if constexpr (PRECONV) {
#if USE_ASYNC
    fill_async(0, 0);
    for (int kt = 0; kt < NT; ++kt) {
      if (kt + 1 < NT) {
        fill_async((kt + 1) & 1, kt + 1);
        __builtin_amdgcn_s_wait_asynccnt(6);   // 6 outstanding = next tile only
      } else {
        __builtin_amdgcn_s_wait_asynccnt(0);
      }
      __syncthreads();           // tile kt visible to all waves
      compute(kt & 1);
      __syncthreads();           // all waves done before buffer reuse
    }
#else
    fill_sync_pre(0, 0);
    __syncthreads();
    for (int kt = 0; kt < NT; ++kt) {
      if (kt + 1 < NT) fill_sync_pre((kt + 1) & 1, kt + 1);
      compute(kt & 1);
      __syncthreads();
    }
#endif
  } else {
    fill_fused(0, 0);
    __syncthreads();
    for (int kt = 0; kt < NT; ++kt) {
      if (kt + 1 < NT) fill_fused((kt + 1) & 1, kt + 1);
      compute(kt & 1);
      __syncthreads();
    }
  }

  // store: C layout -> lanes 0-15 rows M=r, lanes 16-31 rows M=8+r; NT stores
#pragma unroll
  for (int i = 0; i < 4; ++i) {
    const int mbase = m_blk + wm * 64 + i * 16 + (hi << 3);
#pragma unroll
    for (int j = 0; j < 4; ++j) {
      const int n = n_blk + wn * 64 + j * 16 + l15;
#pragma unroll
      for (int r = 0; r < 8; ++r)
        __builtin_nontemporal_store(acc[i][j][r], &out[(size_t)(mbase + r) * N + n]);
    }
  }
}

extern "C" void kernel_launch(void* const* d_in, const int* in_sizes, int n_in,
                              void* d_out, int out_size, void* d_ws, size_t ws_size,
                              hipStream_t stream) {
  const float* x    = (const float*)d_in[0];
  const float* w    = (const float*)d_in[1];
  const float* bias = (const float*)d_in[2];
  float* out = (float*)d_out;

  const long long xs  = in_sizes[0];   // M*K
  const long long wsz = in_sizes[1];   // N*K
  const int N = in_sizes[2];
  const int K = (int)(wsz / N);
  const int M = (int)(xs / K);

  dim3 grid(N / BN, M / BM);
  const size_t need = (size_t)M * K * 2 + (size_t)N * K * 2;

  if (ws_size >= need) {
    unsigned short* xb = (unsigned short*)d_ws;
    unsigned short* wb = xb + (size_t)M * K;
    const long long nx = (long long)M * K, nw = (long long)N * K;
    cvt_x_kernel<<<(unsigned)((nx / 8 + 255) / 256), 256, 0, stream>>>(x, xb, nx);
    cvt_w_kernel<<<(unsigned)((nw / 8 + 255) / 256), 256, 0, stream>>>(w, wb, nw);
    bitnet_gemm<1><<<grid, THREADS, 0, stream>>>(x, w, xb, wb, bias, out, M, N, K);
  } else {
    bitnet_gemm<0><<<grid, THREADS, 0, stream>>>(x, w, nullptr, nullptr, bias, out, M, N, K);
  }
}
